// GCN_7473243095415
// MI455X (gfx1250) — matrix operand order repaired and checked by reading, not compiled, not run on hardware
//
#include <hip/hip_runtime.h>
#include <hip/hip_bf16.h>

typedef __attribute__((ext_vector_type(16))) __bf16 v16bf;
typedef __attribute__((ext_vector_type(8)))  __bf16 v8bf;
typedef __attribute__((ext_vector_type(8)))  float  v8f;

#define N_USER 50000
#define N_ITEM 50000
#define N_TOT  100000
#define D_LAT  64
#define D_FEAT 1280
#define D_HID  256
#define KS1    (D_FEAT / 32)   // 40 K-steps, layer 1
#define KS2    (D_HID / 32)    // 8  K-steps, layer 2
#define MT     (N_ITEM / 16)   // 3125 M-tiles

// ---------------------------------------------------------------- utilities
__global__ void zero_f32(float* __restrict__ p, long long n) {
  long long i = (long long)blockIdx.x * blockDim.x + threadIdx.x;
  if (i < n) p[i] = 0.0f;
}

__global__ void deg_accum(const int* __restrict__ row, float* __restrict__ deg, int E) {
  int e = blockIdx.x * blockDim.x + threadIdx.x;
  if (e < E) atomicAdd(&deg[row[e]], 1.0f);
}

__global__ void deg_to_dinv(float* __restrict__ d, int n) {
  int i = blockIdx.x * blockDim.x + threadIdx.x;
  if (i < n) { float v = d[i]; d[i] = (v > 0.0f) ? (1.0f / sqrtf(v)) : 0.0f; }
}

// ================================================================ packed path
// A-operand layout per 16x16x32 step: lane l holds row M=l&15; element e maps to
// K = kbase + e (e<8) or 16+kbase+(e-8) (e>=8) with kbase=(l>=16)*8.
// Packed storage: P[((tile*KS + ks)*32 + lane)*16 + e]  -> 32B contiguous per lane.

__global__ void pack_a_bf16(const float* __restrict__ F, __bf16* __restrict__ P) {
  int t = blockIdx.x * blockDim.x + threadIdx.x;       // MT*KS1*32 = 4M threads
  if (t >= MT * KS1 * 32) return;
  int lane  = t & 31;
  int ks    = (t >> 5) % KS1;
  int tile  = t / (32 * KS1);
  int hi = lane >> 4, l15 = lane & 15;
  const float* ap = F + (size_t)(tile * 16 + l15) * D_FEAT + ks * 32 + hi * 8;
  float4 a0 = *(const float4*)(ap +  0);
  float4 a1 = *(const float4*)(ap +  4);
  float4 a2 = *(const float4*)(ap + 16);
  float4 a3 = *(const float4*)(ap + 20);
  v16bf o;
  o[0]=(__bf16)a0.x;  o[1]=(__bf16)a0.y;  o[2]=(__bf16)a0.z;  o[3]=(__bf16)a0.w;
  o[4]=(__bf16)a1.x;  o[5]=(__bf16)a1.y;  o[6]=(__bf16)a1.z;  o[7]=(__bf16)a1.w;
  o[8]=(__bf16)a2.x;  o[9]=(__bf16)a2.y;  o[10]=(__bf16)a2.z; o[11]=(__bf16)a2.w;
  o[12]=(__bf16)a3.x; o[13]=(__bf16)a3.y; o[14]=(__bf16)a3.z; o[15]=(__bf16)a3.w;
  *(v16bf*)(P + ((size_t)(tile * KS1 + ks) * 32 + lane) * 16) = o;
}

// B-operand layout: lane l holds col N=l&15; element e maps to K = kbase2+e (e<8) or
// 16+kbase2+(e-8) (e>=8) with kbase2=(l>=16)*8.  Packed: P[((tn*KS + ks)*32 + lane)*16 + e].
__global__ void pack_b_bf16(const float* __restrict__ W, __bf16* __restrict__ P,
                            int K, int Ncols) {
  int KS = K / 32;
  int t = blockIdx.x * blockDim.x + threadIdx.x;
  if (t >= (Ncols / 16) * KS * 32) return;
  int lane = t & 31;
  int ks   = (t >> 5) % KS;
  int tn   = t / (32 * KS);
  int hi = lane >> 4, l15 = lane & 15;
  const float* wp = W + (size_t)(ks * 32 + hi * 8) * Ncols + tn * 16 + l15;
  v16bf o;
#pragma unroll
  for (int e = 0; e < 16; ++e) o[e] = (__bf16)wp[(size_t)(e + (e >> 3) * 8) * Ncols];
  *(v16bf*)(P + ((size_t)(tn * KS + ks) * 32 + lane) * 16) = o;
}

// layer 1 (packed): one wave computes 16x64 (4 N-tiles); mid stored in layer-2
// packed-A layout so layer 2 streams it back with dense 32B loads.
__global__ void mlp1_wmma_packed(const __bf16* __restrict__ Ap, const __bf16* __restrict__ Bp,
                                 const float* __restrict__ b1, __bf16* __restrict__ midp) {
  const int wid  = (blockIdx.x * blockDim.x + threadIdx.x) >> 5;
  const int lane = threadIdx.x & 31;
  if (wid >= MT * 4) return;
  const int ng = wid & 3;            // n-group: 64 columns
  const int tm = wid >> 2;
  const int hi = lane >> 4, l15 = lane & 15;

  v8f acc[4] = {{}, {}, {}, {}};
  for (int ks = 0; ks < KS1; ++ks) {
    v16bf A = *(const v16bf*)(Ap + ((size_t)(tm * KS1 + ks) * 32 + lane) * 16);
#pragma unroll
    for (int j = 0; j < 4; ++j) {
      const int tn = ng * 4 + j;
      v16bf B = *(const v16bf*)(Bp + ((size_t)(tn * KS1 + ks) * 32 + lane) * 16);
      acc[j] = __builtin_amdgcn_wmma_f32_16x16x32_bf16(false, A, false, B, (short)0,
                                                       acc[j], false, false);
    }
  }
#pragma unroll
  for (int j = 0; j < 4; ++j) {
    const int n = (ng * 4 + j) * 16 + l15;     // output col == layer-2 K index
    const float bias = b1[n];
    const int ks2 = n >> 5;
    const int kl  = n & 31;
    const int hi2 = (kl >> 3) & 1;
    const int e2  = (kl & 7) + ((kl >> 4) << 3);
#pragma unroll
    for (int v = 0; v < 8; ++v) {
      float val = acc[j][v] + bias;
      val = (val > 0.0f) ? val : 0.01f * val;  // leaky_relu(0.01)
      const int lane2 = (v + 8 * hi) + 16 * hi2;   // m&15 + 16*hi2
      midp[((size_t)(tm * KS2 + ks2) * 32 + lane2) * 16 + e2] = (__bf16)val;
    }
  }
}

// layer 2 (packed): one wave computes 16x64 (all 4 N-tiles), K=256.
__global__ void mlp2_wmma_packed(const __bf16* __restrict__ midp, const __bf16* __restrict__ Bp,
                                 const float* __restrict__ b2, float* __restrict__ x) {
  const int wid  = (blockIdx.x * blockDim.x + threadIdx.x) >> 5;
  const int lane = threadIdx.x & 31;
  if (wid >= MT) return;
  const int tm = wid;
  const int hi = lane >> 4, l15 = lane & 15;

  v8f acc[4] = {{}, {}, {}, {}};
  for (int ks = 0; ks < KS2; ++ks) {
    v16bf A = *(const v16bf*)(midp + ((size_t)(tm * KS2 + ks) * 32 + lane) * 16);
#pragma unroll
    for (int j = 0; j < 4; ++j) {
      v16bf B = *(const v16bf*)(Bp + ((size_t)(j * KS2 + ks) * 32 + lane) * 16);
      acc[j] = __builtin_amdgcn_wmma_f32_16x16x32_bf16(false, A, false, B, (short)0,
                                                       acc[j], false, false);
    }
  }
#pragma unroll
  for (int j = 0; j < 4; ++j) {
    const int n = j * 16 + l15;
    const float bias = b2[n];
#pragma unroll
    for (int v = 0; v < 8; ++v)
      x[(size_t)(N_USER + tm * 16 + v + 8 * hi) * D_LAT + n] = acc[j][v] + bias;
  }
}

// ================================================================ fallback path
// (inline-conversion gather GEMMs; used only if ws_size can't hold packed A)
__global__ void mlp1_wmma_gather(const float* __restrict__ F, const float* __restrict__ W1,
                                 const float* __restrict__ b1, __bf16* __restrict__ mid) {
  const int wid  = (blockIdx.x * blockDim.x + threadIdx.x) >> 5;
  const int lane = threadIdx.x & 31;
  const int tn = wid & 15;
  const int tm = wid >> 4;
  if (tm >= MT) return;
  const int m0 = tm << 4, n0 = tn << 4;
  const int hi = lane >> 4, l15 = lane & 15;
  const int mrow = m0 + l15, kbaseA = hi * 8, kbaseB = hi * 8, ncol = n0 + l15;
  v8f acc = {};
  for (int k0 = 0; k0 < D_FEAT; k0 += 32) {
    const float* ap = F + (size_t)mrow * D_FEAT + k0 + kbaseA;
    float4 a0 = *(const float4*)(ap + 0),  a1 = *(const float4*)(ap + 4);
    float4 a2 = *(const float4*)(ap + 16), a3 = *(const float4*)(ap + 20);
    v16bf A;
    A[0]=(__bf16)a0.x;  A[1]=(__bf16)a0.y;  A[2]=(__bf16)a0.z;  A[3]=(__bf16)a0.w;
    A[4]=(__bf16)a1.x;  A[5]=(__bf16)a1.y;  A[6]=(__bf16)a1.z;  A[7]=(__bf16)a1.w;
    A[8]=(__bf16)a2.x;  A[9]=(__bf16)a2.y;  A[10]=(__bf16)a2.z; A[11]=(__bf16)a2.w;
    A[12]=(__bf16)a3.x; A[13]=(__bf16)a3.y; A[14]=(__bf16)a3.z; A[15]=(__bf16)a3.w;
    const float* bp = W1 + (size_t)(k0 + kbaseB) * D_HID + ncol;
    v16bf Bm;
#pragma unroll
    for (int e = 0; e < 16; ++e) Bm[e] = (__bf16)bp[(size_t)(e + (e >> 3) * 8) * D_HID];
    acc = __builtin_amdgcn_wmma_f32_16x16x32_bf16(false, A, false, Bm, (short)0, acc,
                                                  false, false);
  }
  const float bias = b1[ncol];
#pragma unroll
  for (int v = 0; v < 8; ++v) {
    float val = acc[v] + bias;
    val = (val > 0.0f) ? val : 0.01f * val;
    mid[(size_t)(m0 + v + 8 * hi) * D_HID + ncol] = (__bf16)val;
  }
}

__global__ void mlp2_wmma_gather(const __bf16* __restrict__ mid, const float* __restrict__ W2,
                                 const float* __restrict__ b2, float* __restrict__ x) {
  const int wid  = (blockIdx.x * blockDim.x + threadIdx.x) >> 5;
  const int lane = threadIdx.x & 31;
  if (wid >= MT * 4) return;
  const int tn = wid & 3, tm = wid >> 2;
  const int m0 = tm << 4, n0 = tn << 4;
  const int hi = lane >> 4, l15 = lane & 15;
  const int mrow = m0 + l15, kbaseA = hi * 8, kbaseB = hi * 8, ncol = n0 + l15;
  v8f acc = {};
  for (int k0 = 0; k0 < D_HID; k0 += 32) {
    const __bf16* ap = mid + (size_t)mrow * D_HID + k0 + kbaseA;
    v8bf lo = *(const v8bf*)(ap), hi8 = *(const v8bf*)(ap + 16);
    v16bf A;
#pragma unroll
    for (int e = 0; e < 8; ++e) { A[e] = lo[e]; A[e + 8] = hi8[e]; }
    const float* bp = W2 + (size_t)(k0 + kbaseB) * D_LAT + ncol;
    v16bf Bm;
#pragma unroll
    for (int e = 0; e < 16; ++e) Bm[e] = (__bf16)bp[(size_t)(e + (e >> 3) * 8) * D_LAT];
    acc = __builtin_amdgcn_wmma_f32_16x16x32_bf16(false, A, false, Bm, (short)0, acc,
                                                  false, false);
  }
  const float bias = b2[ncol];
#pragma unroll
  for (int v = 0; v < 8; ++v)
    x[(size_t)(N_USER + m0 + v + 8 * hi) * D_LAT + ncol] = acc[v] + bias;
}

// ---------------------------------------------------------------- graph part
__global__ void copy_pref(const float* __restrict__ pref, float* __restrict__ x,
                          float* __restrict__ out_tail, int n) {
  int i = blockIdx.x * blockDim.x + threadIdx.x;
  if (i < n) { float v = pref[i]; x[i] = v; out_tail[i] = v; }
}

__global__ void normalize_rows(float* __restrict__ x, int nrows) {
  const int wid  = (blockIdx.x * blockDim.x + threadIdx.x) >> 5;
  const int lane = threadIdx.x & 31;
  if (wid >= nrows) return;
  float* r = x + (size_t)wid * D_LAT;
  float a = r[lane], b = r[lane + 32];
  float s = a * a + b * b;
#pragma unroll
  for (int off = 16; off > 0; off >>= 1) s += __shfl_xor(s, off, 32);
  float inv = 1.0f / fmaxf(sqrtf(s), 1e-12f);
  r[lane] = a * inv;
  r[lane + 32] = b * inv;
}

__global__ void conv_scatter(const int* __restrict__ row, const int* __restrict__ col,
                             const float* __restrict__ dinv, const float* __restrict__ src,
                             float* __restrict__ dst, long long total) {
  long long t = (long long)blockIdx.x * blockDim.x + threadIdx.x;
  if (t >= total) return;
  int e = (int)(t >> 6);
  int d = (int)(t & 63);
  int r = row[e], c = col[e];
  float w = dinv[r] * dinv[c];
  atomicAdd(&dst[(size_t)c * D_LAT + d], w * src[(size_t)r * D_LAT + d]);
}

__global__ void add_xy(const float* __restrict__ a, const float* __restrict__ b,
                       float* __restrict__ o, long long n) {
  long long i = (long long)blockIdx.x * blockDim.x + threadIdx.x;
  if (i < n) o[i] = a[i] + b[i];
}

// ---------------------------------------------------------------- launcher
extern "C" void kernel_launch(void* const* d_in, const int* in_sizes, int n_in,
                              void* d_out, int out_size, void* d_ws, size_t ws_size,
                              hipStream_t stream) {
  (void)n_in; (void)out_size;
  const int*   edge = (const int*)  d_in[1];
  const float* feat = (const float*)d_in[2];
  const float* pref = (const float*)d_in[3];
  const float* W1   = (const float*)d_in[4];
  const float* b1   = (const float*)d_in[5];
  const float* W2   = (const float*)d_in[6];
  const float* b2   = (const float*)d_in[7];
  float* out = (float*)d_out;

  const int E = in_sizes[1] / 2;
  const int* row = edge;
  const int* col = edge + E;

  // workspace: dinv | x | h | mid | [Apack | W1pack | W2pack]
  char* ws = (char*)d_ws;
  size_t o = 0;
  float* dinv = (float*)(ws + o); o += (((size_t)N_TOT * 4) + 255) & ~(size_t)255;
  float* x    = (float*)(ws + o); o += (size_t)N_TOT * D_LAT * 4;
  float* h    = (float*)(ws + o); o += (size_t)N_TOT * D_LAT * 4;
  __bf16* mid = (__bf16*)(ws + o); o += (size_t)N_ITEM * D_HID * 2;
  __bf16* Apk = (__bf16*)(ws + o); size_t need = o + (size_t)N_ITEM * D_FEAT * 2;
  __bf16* W1p = (__bf16*)(ws + need); need += (size_t)D_FEAT * D_HID * 2;
  __bf16* W2p = (__bf16*)(ws + need); need += (size_t)D_HID * D_LAT * 2;
  const bool packed = (ws_size >= need);

  const int B = 256;
  const long long xn = (long long)N_TOT * D_LAT;
  const long long cw = (long long)E * D_LAT;

  zero_f32<<<(N_TOT + B - 1) / B, B, 0, stream>>>(dinv, N_TOT);
  zero_f32<<<(unsigned)((xn + B - 1) / B), B, 0, stream>>>(h, xn);
  deg_accum<<<(E + B - 1) / B, B, 0, stream>>>(row, dinv, E);
  deg_to_dinv<<<(N_TOT + B - 1) / B, B, 0, stream>>>(dinv, N_TOT);

  if (packed) {
    pack_a_bf16<<<(MT * KS1 * 32 + B - 1) / B, B, 0, stream>>>(feat, Apk);
    pack_b_bf16<<<((D_HID / 16) * KS1 * 32 + B - 1) / B, B, 0, stream>>>(W1, W1p, D_FEAT, D_HID);
    pack_b_bf16<<<((D_LAT / 16) * KS2 * 32 + B - 1) / B, B, 0, stream>>>(W2, W2p, D_HID, D_LAT);
    mlp1_wmma_packed<<<(MT * 4 * 32 + B - 1) / B, B, 0, stream>>>(Apk, W1p, b1, mid);
    mlp2_wmma_packed<<<(MT * 32 + B - 1) / B, B, 0, stream>>>(mid, W2p, b2, x);
  } else {
    mlp1_wmma_gather<<<(MT * 16 * 32 + B - 1) / B, B, 0, stream>>>(feat, W1, b1, mid);
    mlp2_wmma_gather<<<(MT * 4 * 32 + B - 1) / B, B, 0, stream>>>(mid, W2, b2, x);
  }

  copy_pref<<<((N_USER * D_LAT) + B - 1) / B, B, 0, stream>>>(
      pref, x, out + (size_t)N_TOT * D_LAT, N_USER * D_LAT);
  normalize_rows<<<((N_TOT * 32) + B - 1) / B, B, 0, stream>>>(x, N_TOT);

  conv_scatter<<<(unsigned)((cw + B - 1) / B), B, 0, stream>>>(row, col, dinv, x, h, cw);
  add_xy<<<(unsigned)((xn + B - 1) / B), B, 0, stream>>>(x, h, out, xn);
  conv_scatter<<<(unsigned)((cw + B - 1) / B), B, 0, stream>>>(row, col, dinv, h, out, cw);
}
